// CrossAttention_Epipolar_35888746725715
// MI455X (gfx1250) — matrix operand order repaired and checked by reading, not compiled, not run on hardware
//
#include <hip/hip_runtime.h>
#include <hip/hip_bf16.h>
#include <math.h>

#define HEADS  16
#define DH     64
#define TT     77     // text keys
#define TTP    96     // text keys padded (multiple of 32 for K-tiling)
#define TI     256    // image keys
#define CTXLEN 333
#define BATCH  32
#define NQ     1024
#define DM     1024

typedef __attribute__((ext_vector_type(16))) _Float16 v16h;
typedef __attribute__((ext_vector_type(8)))  float    v8f;

union F16Frag { v16h v; uint4 u[2]; };

__device__ __forceinline__ v8f vzero8() {
  v8f c;
#pragma unroll
  for (int r = 0; r < 8; ++r) c[r] = 0.f;
  return c;
}

__device__ __forceinline__ v8f wmma_f16(v16h a, v16h b, v8f c) {
  // v_wmma_f32_16x16x32_f16: (neg_a, A, neg_b, B, c_mod, C, reuse_a, reuse_b)
  return __builtin_amdgcn_wmma_f32_16x16x32_f16(false, a, false, b, (short)0, c, false, false);
}

// ---- per-lane-pointer fragment loaders (row pointer precomputed once) ----
// A row ptr = base + lm*lda + hh*8 ; fragment at K offset k0
__device__ __forceinline__ v16h lda_p(const _Float16* ap, int k0) {
  F16Frag f;
  f.u[0] = *(const uint4*)(ap + k0);
  f.u[1] = *(const uint4*)(ap + k0 + 16);
  return f.v;
}
// B row ptr = base + lm*ldb + hh*8 ; fragment at K offset k0
__device__ __forceinline__ v16h ldb_p(const _Float16* bp, int k0) {
  F16Frag f;
  f.u[0] = *(const uint4*)(bp + k0);
  f.u[1] = *(const uint4*)(bp + k0 + 16);
  return f.v;
}

// ---- generic fragment loaders (used by the attention kernel) ----
// A-matrix 16x32 f16 fragment (ISA 7.12.2): lane<16 row=lm holds K=k0..k0+7 , k0+16..k0+23
//                                           lane>=16 row=lm holds K=k0+8..k0+15, k0+24..k0+31
__device__ __forceinline__ v16h ld_a(const _Float16* base, int lda, int k0, int lane) {
  const int hh = lane >> 4, lm = lane & 15;
  const _Float16* p = base + lm * lda + k0 + hh * 8;
  F16Frag f;
  f.u[0] = *(const uint4*)(p);
  f.u[1] = *(const uint4*)(p + 16);
  return f.v;
}
// B-matrix 32x16 f16 fragment from a transposed [N][K] buffer:
// lane<16 col n=lm holds K=k0..k0+7, k0+16..k0+23 ; lane>=16 col n=lm holds K=k0+8..k0+15, k0+24..k0+31
__device__ __forceinline__ v16h ld_b(const _Float16* bt, int ldb, int k0, int lane) {
  const int hh = lane >> 4, lm = lane & 15;
  const _Float16* p = bt + lm * ldb + k0 + hh * 8;
  F16Frag f;
  f.u[0] = *(const uint4*)(p);
  f.u[1] = *(const uint4*)(p + 16);
  return f.v;
}

// A fragment gathered from an f32 LDS buffer (row-major, stride lda floats)
__device__ __forceinline__ v16h ld_a_f32(const float* base, int lda, int k0, int lane) {
  const int hh = lane >> 4, lm = lane & 15;
  v16h r;
#pragma unroll
  for (int e = 0; e < 8; ++e) r[e]     = (_Float16)base[lm * lda + k0 + hh * 8 + e];
#pragma unroll
  for (int e = 0; e < 8; ++e) r[8 + e] = (_Float16)base[lm * lda + k0 + 16 + hh * 8 + e];
  return r;
}

// ---------------- elementwise / prep kernels ----------------

__global__ void cvt_f16(const float* __restrict__ in, _Float16* __restrict__ out, int n) {
  int id = blockIdx.x * 256 + threadIdx.x;
  if (id < n) out[id] = (_Float16)in[id];
}

// Wt[n][k] = (f16) W[k][n]   (1024x1024)
__global__ void transpose_w16(const float* __restrict__ W, _Float16* __restrict__ Wt) {
  int id = blockIdx.x * 256 + threadIdx.x;   // 0 .. 1M-1
  int nrow = id >> 10, k = id & 1023;
  Wt[id] = (_Float16)W[k * 1024 + nrow];
}

// relk16: [48][64]  (B^T layout for N=t, K=d), zero-padded t>=33
// relvT16:[64][64]  (B^T layout for N=d, K=t), zero-padded t>=33
__global__ void prep_rel(const float* __restrict__ rel_k, const float* __restrict__ rel_v,
                         _Float16* __restrict__ relk16, _Float16* __restrict__ relvT16) {
  int id = blockIdx.x * 256 + threadIdx.x;   // 0..4095
  if (id < 48 * 64) {
    int t = id >> 6, d = id & 63;
    relk16[id] = (t < 33) ? (_Float16)rel_k[t * 64 + d] : (_Float16)0.f;
  }
  {
    int d = id >> 6, t = id & 63;
    relvT16[id] = (t < 33) ? (_Float16)rel_v[t * 64 + d] : (_Float16)0.f;
  }
}

// ---------------- WMMA GEMM core: 16M x 64N per wave ----------------
// Ping-pong double buffering, unrolled x2 so no register-rotation copies are needed.
__device__ __forceinline__ void gemm_16x64(const _Float16* ap, const _Float16* bp, v8f acc[4]) {
  const _Float16* bp1 = bp + (size_t)16 * DM;
  const _Float16* bp2 = bp + (size_t)32 * DM;
  const _Float16* bp3 = bp + (size_t)48 * DM;

  v16h aA  = lda_p(ap, 0);
  v16h bA0 = ldb_p(bp, 0),  bA1 = ldb_p(bp1, 0);
  v16h bA2 = ldb_p(bp2, 0), bA3 = ldb_p(bp3, 0);
  v16h aB, bB0, bB1, bB2, bB3;

  int k0 = 0;
  for (; k0 < DM - 64; k0 += 64) {
    // phase B loads (k0+32), then consume phase A (k0)
    aB  = lda_p(ap, k0 + 32);
    bB0 = ldb_p(bp,  k0 + 32); bB1 = ldb_p(bp1, k0 + 32);
    bB2 = ldb_p(bp2, k0 + 32); bB3 = ldb_p(bp3, k0 + 32);
    acc[0] = wmma_f16(aA, bA0, acc[0]);
    acc[1] = wmma_f16(aA, bA1, acc[1]);
    acc[2] = wmma_f16(aA, bA2, acc[2]);
    acc[3] = wmma_f16(aA, bA3, acc[3]);
    // phase A loads (k0+64), then consume phase B (k0+32)
    aA  = lda_p(ap, k0 + 64);
    bA0 = ldb_p(bp,  k0 + 64); bA1 = ldb_p(bp1, k0 + 64);
    bA2 = ldb_p(bp2, k0 + 64); bA3 = ldb_p(bp3, k0 + 64);
    acc[0] = wmma_f16(aB, bB0, acc[0]);
    acc[1] = wmma_f16(aB, bB1, acc[1]);
    acc[2] = wmma_f16(aB, bB2, acc[2]);
    acc[3] = wmma_f16(aB, bB3, acc[3]);
  }
  // epilogue: k0 == DM-64, consume last two fragment sets
  aB  = lda_p(ap, k0 + 32);
  bB0 = ldb_p(bp,  k0 + 32); bB1 = ldb_p(bp1, k0 + 32);
  bB2 = ldb_p(bp2, k0 + 32); bB3 = ldb_p(bp3, k0 + 32);
  acc[0] = wmma_f16(aA, bA0, acc[0]);
  acc[1] = wmma_f16(aA, bA1, acc[1]);
  acc[2] = wmma_f16(aA, bA2, acc[2]);
  acc[3] = wmma_f16(aA, bA3, acc[3]);
  acc[0] = wmma_f16(aB, bB0, acc[0]);
  acc[1] = wmma_f16(aB, bB1, acc[1]);
  acc[2] = wmma_f16(aB, bB2, acc[2]);
  acc[3] = wmma_f16(aB, bB3, acc[3]);
}

// Q = x @ Wq, scaled by DIM_HEAD^-0.5, stored per-head [b][h][i][d]
// grid (2048, 2) block (32,8): column group ng = 0..15 covers one head (64 cols)
__global__ void __launch_bounds__(256) proj_q(const _Float16* __restrict__ x16,
                                              const _Float16* __restrict__ Wt,
                                              _Float16* __restrict__ Q16) {
  const int lane = threadIdx.x;
  const int hh = lane >> 4, lm = lane & 15;
  const int mt = blockIdx.x;                       // 0..2047
  const int ng = blockIdx.y * 8 + threadIdx.y;     // 0..15 (head)
  const _Float16* ap = x16 + ((size_t)mt * 16 + lm) * DM + hh * 8;
  const _Float16* bp = Wt + ((size_t)ng * 64 + lm) * DM + hh * 8;
  v8f acc[4];
#pragma unroll
  for (int t = 0; t < 4; ++t) acc[t] = vzero8();
  gemm_16x64(ap, bp, acc);
#pragma unroll
  for (int t = 0; t < 4; ++t) {
    const int d = t * 16 + lm;                     // h == ng
#pragma unroll
    for (int r = 0; r < 8; ++r) {
      int row = mt * 16 + hh * 8 + r;
      int b = row >> 10, i = row & 1023;
      Q16[(((size_t)(b * HEADS + ng)) * NQ + i) * DH + d] = (_Float16)(acc[t][r] * 0.125f);
    }
  }
}

// context projection: K (normal per-head [b][h][j][d]) or V (transposed [b][h][d][j])
// grid (Tpad/16, 2, B) block (32,8).  Padded token rows use a clamped source row
// (finite garbage) -- provably masked downstream (sim mask + zeroed attn columns).
__global__ void __launch_bounds__(256) proj_ctx(const _Float16* __restrict__ ctx16,
                                                int row_off, int T, int Tpad,
                                                const _Float16* __restrict__ Wt,
                                                _Float16* __restrict__ out, int transposed) {
  const int lane = threadIdx.x;
  const int hh = lane >> 4, lm = lane & 15;
  const int mt = blockIdx.x;                       // token tile
  const int ng = blockIdx.y * 8 + threadIdx.y;     // 0..15 (head)
  const int b  = blockIdx.z;
  const int srcRow = min(mt * 16 + lm, T - 1);     // clamp padded rows
  const _Float16* ap = ctx16 + ((size_t)(b * CTXLEN + row_off) + srcRow) * DM + hh * 8;
  const _Float16* bp = Wt + ((size_t)ng * 64 + lm) * DM + hh * 8;
  v8f acc[4];
#pragma unroll
  for (int t = 0; t < 4; ++t) acc[t] = vzero8();
  gemm_16x64(ap, bp, acc);
  const size_t bhBase = (size_t)(b * HEADS + ng);
#pragma unroll
  for (int t = 0; t < 4; ++t) {
    const int d = t * 16 + lm;
#pragma unroll
    for (int r = 0; r < 8; ++r) {
      int j = mt * 16 + hh * 8 + r;
      if (!transposed)
        out[(bhBase * Tpad + j) * DH + d] = (_Float16)acc[t][r];
      else
        out[(bhBase * DH + d) * Tpad + j] = (_Float16)acc[t][r];
    }
  }
}

// y = Ob @ Wout + bout  (f32 output);  grid (2048, 2) block (32,8)
__global__ void __launch_bounds__(256) out_proj(const _Float16* __restrict__ Ob,
                                                const _Float16* __restrict__ Wt,
                                                const float* __restrict__ bout,
                                                float* __restrict__ y) {
  const int lane = threadIdx.x;
  const int hh = lane >> 4, lm = lane & 15;
  const int mt = blockIdx.x;
  const int ng = blockIdx.y * 8 + threadIdx.y;
  const _Float16* ap = Ob + ((size_t)mt * 16 + lm) * DM + hh * 8;
  const _Float16* bp = Wt + ((size_t)ng * 64 + lm) * DM + hh * 8;
  v8f acc[4];
#pragma unroll
  for (int t = 0; t < 4; ++t) acc[t] = vzero8();
  gemm_16x64(ap, bp, acc);
#pragma unroll
  for (int t = 0; t < 4; ++t) {
    const int col = ng * 64 + t * 16 + lm;
    const float bv = bout[col];
#pragma unroll
    for (int r = 0; r < 8; ++r) {
      int row = mt * 16 + hh * 8 + r;
      y[(size_t)row * DM + col] = acc[t][r] + bv;
    }
  }
}

// ---------------- fused attention kernel ----------------
// one wave (32 threads) per (b, h, 16-query tile)
__global__ void __launch_bounds__(32) attn_kernel(
    const _Float16* __restrict__ Q16, const _Float16* __restrict__ K16,
    const _Float16* __restrict__ Vt16, const _Float16* __restrict__ Kip16,
    const _Float16* __restrict__ VipT16, const _Float16* __restrict__ relk16,
    const _Float16* __restrict__ relvT16, const float* __restrict__ alpha,
    _Float16* __restrict__ Ob) {
  __shared__ __attribute__((aligned(16))) float    lds_qrel[16][48];
  __shared__ __attribute__((aligned(16))) float    lds_w[16][64];
  __shared__ __attribute__((aligned(16))) _Float16 lds_attn[16][TTP];
  __shared__ __attribute__((aligned(16))) float    lds_simip[16][TI];
  __shared__ __attribute__((aligned(16))) _Float16 lds_attn_ip[16][TI];

  const int lane = threadIdx.x;
  const int hh = lane >> 4, lm = lane & 15;
  const int qt = blockIdx.x, h = blockIdx.y, b = blockIdx.z;
  const int bh = b * HEADS + h;

  for (int e = lane; e < 16 * 64; e += 32)  (&lds_w[0][0])[e] = 0.f;
  for (int e = lane; e < 16 * TTP; e += 32) (&lds_attn[0][0])[e] = (_Float16)0.f;

  // Q A-fragments (K=64 -> two 16x32 fragments); SCALE baked in at proj_q
  const _Float16* qb = Q16 + ((size_t)bh * NQ + qt * 16) * DH;
  const v16h a0 = ld_a(qb, DH, 0, lane);
  const v16h a1 = ld_a(qb, DH, 32, lane);

  // ---- text sim = q @ k^T (77 keys; padded rows masked below) ----
  const _Float16* kb = K16 + (size_t)bh * TTP * DH;
  v8f s[5];
#pragma unroll
  for (int jt = 0; jt < 5; ++jt) {
    v8f c = vzero8();
    c = wmma_f16(a0, ld_b(kb + jt * 16 * DH, DH, 0, lane), c);
    c = wmma_f16(a1, ld_b(kb + jt * 16 * DH, DH, 32, lane), c);
    s[jt] = c;
  }

  // ---- qrel[m][t] = q[m] . rel_k[t] (t padded to 48) ----
#pragma unroll
  for (int tt = 0; tt < 3; ++tt) {
    v8f c = vzero8();
    c = wmma_f16(a0, ld_b(relk16 + tt * 16 * DH, DH, 0, lane), c);
    c = wmma_f16(a1, ld_b(relk16 + tt * 16 * DH, DH, 32, lane), c);
#pragma unroll
    for (int r = 0; r < 8; ++r) lds_qrel[hh * 8 + r][tt * 16 + lm] = c[r];
  }
  __syncthreads();

  // ---- text softmax (rows spread over 16-lane groups) ----
  float vals[5][8];
#pragma unroll
  for (int jt = 0; jt < 5; ++jt)
#pragma unroll
    for (int r = 0; r < 8; ++r) {
      const int j = jt * 16 + lm, m = hh * 8 + r;
      float v = s[jt][r];
      if (j < TT) {
        const int i = qt * 16 + m;
        const int t = min(max(j - i, -16), 16) + 16;
        v += lds_qrel[m][t];
      } else v = -1e30f;
      vals[jt][r] = v;
    }
#pragma unroll
  for (int r = 0; r < 8; ++r) {
    float mx = vals[0][r];
#pragma unroll
    for (int jt = 1; jt < 5; ++jt) mx = fmaxf(mx, vals[jt][r]);
    for (int msk = 1; msk < 16; msk <<= 1) mx = fmaxf(mx, __shfl_xor(mx, msk));
    float e[5], sm = 0.f;
#pragma unroll
    for (int jt = 0; jt < 5; ++jt) { e[jt] = __expf(vals[jt][r] - mx); sm += e[jt]; }
    for (int msk = 1; msk < 16; msk <<= 1) sm += __shfl_xor(sm, msk);
    const float inv = 1.f / sm;
    const int m = hh * 8 + r;
#pragma unroll
    for (int jt = 0; jt < 5; ++jt) {
      const int j = jt * 16 + lm;
      if (j < TT) {
        const float p = e[jt] * inv;
        lds_attn[m][j] = (_Float16)p;
        const int i = qt * 16 + m;
        const int t = min(max(j - i, -16), 16) + 16;
        atomicAdd(&lds_w[m][t], p);      // ds_add_f32 bin-sum for rel_v
      }
    }
  }
  __syncthreads();

  // ---- text out = attn @ v^T + w @ rel_v^T ----
  v8f o[4];
  const _Float16* vtb = Vt16 + (size_t)bh * DH * TTP;
#pragma unroll
  for (int dt = 0; dt < 4; ++dt) {
    v8f c = vzero8();
#pragma unroll
    for (int k0 = 0; k0 < TTP; k0 += 32)
      c = wmma_f16(ld_a(&lds_attn[0][0], TTP, k0, lane),
                   ld_b(vtb + dt * 16 * TTP, TTP, k0, lane), c);
#pragma unroll
    for (int k0 = 0; k0 < 64; k0 += 32)
      c = wmma_f16(ld_a_f32(&lds_w[0][0], 64, k0, lane),
                   ld_b(relvT16 + dt * 16 * 64, 64, k0, lane), c);
    o[dt] = c;
  }

  // ---- image branch: sim_ip (16x256) staged to LDS ----
  const _Float16* kipb = Kip16 + (size_t)bh * TI * DH;
  for (int jt = 0; jt < 16; ++jt) {
    v8f c = vzero8();
    c = wmma_f16(a0, ld_b(kipb + jt * 16 * DH, DH, 0, lane), c);
    c = wmma_f16(a1, ld_b(kipb + jt * 16 * DH, DH, 32, lane), c);
#pragma unroll
    for (int r = 0; r < 8; ++r) lds_simip[hh * 8 + r][jt * 16 + lm] = c[r];
  }
  __syncthreads();
  {
    const int row = lm;                    // half-wave splits columns
    float mx = -1e30f;
    for (int c0 = 0; c0 < 128; ++c0) mx = fmaxf(mx, lds_simip[row][hh * 128 + c0]);
    mx = fmaxf(mx, __shfl_xor(mx, 16));
    float sm = 0.f;
    for (int c0 = 0; c0 < 128; ++c0) sm += __expf(lds_simip[row][hh * 128 + c0] - mx);
    sm += __shfl_xor(sm, 16);
    const float inv = 1.f / sm;
    for (int c0 = 0; c0 < 128; ++c0) {
      const int cc = hh * 128 + c0;
      lds_attn_ip[row][cc] = (_Float16)(__expf(lds_simip[row][cc] - mx) * inv);
    }
  }
  __syncthreads();

  // ---- out_ip = attn_ip @ v_ip^T ; merge with gate ----
  const _Float16* vipb = VipT16 + (size_t)bh * DH * TI;
  const float g = tanhf(alpha[0]) + 1.0f;   // IC_SCALE == 1
#pragma unroll
  for (int dt = 0; dt < 4; ++dt) {
    v8f c = vzero8();
    for (int k0 = 0; k0 < TI; k0 += 32)
      c = wmma_f16(ld_a(&lds_attn_ip[0][0], TI, k0, lane),
                   ld_b(vipb + dt * 16 * TI, TI, k0, lane), c);
    const int col = h * DH + dt * 16 + lm;
#pragma unroll
    for (int r = 0; r < 8; ++r) {
      const int i = qt * 16 + hh * 8 + r;
      Ob[((size_t)(b * NQ + i)) * DM + col] = (_Float16)(o[dt][r] + g * c[r]);
    }
  }
}

// ---------------- host launcher ----------------

extern "C" void kernel_launch(void* const* d_in, const int* in_sizes, int n_in,
                              void* d_out, int out_size, void* d_ws, size_t ws_size,
                              hipStream_t stream) {
  (void)in_sizes; (void)n_in; (void)out_size; (void)ws_size;
  const float* x      = (const float*)d_in[0];
  const float* ctx    = (const float*)d_in[1];
  const float* Wq     = (const float*)d_in[2];
  const float* Wk     = (const float*)d_in[3];
  const float* Wv     = (const float*)d_in[4];
  const float* Wk_ip  = (const float*)d_in[5];
  const float* Wv_ip  = (const float*)d_in[6];
  const float* rel_k  = (const float*)d_in[7];
  const float* rel_v  = (const float*)d_in[8];
  const float* alpha  = (const float*)d_in[9];
  const float* Wout   = (const float*)d_in[10];
  const float* bout   = (const float*)d_in[11];
  float* y = (float*)d_out;

  char* ws = (char*)d_ws;
  size_t off = 0;
  auto take = [&](size_t bytes) -> char* {
    char* p = ws + off;
    off += (bytes + 255) & ~(size_t)255;
    return p;
  };
  _Float16* xob16  = (_Float16*)take((size_t)BATCH * NQ * DM * 2);   // x f16, reused as merged out
  _Float16* ctx16  = (_Float16*)take((size_t)BATCH * CTXLEN * DM * 2);
  _Float16* WqT    = (_Float16*)take((size_t)DM * DM * 2);
  _Float16* WkT    = (_Float16*)take((size_t)DM * DM * 2);
  _Float16* WvT    = (_Float16*)take((size_t)DM * DM * 2);
  _Float16* WkipT  = (_Float16*)take((size_t)DM * DM * 2);
  _Float16* WvipT  = (_Float16*)take((size_t)DM * DM * 2);
  _Float16* WoT    = (_Float16*)take((size_t)DM * DM * 2);
  _Float16* Q16    = (_Float16*)take((size_t)BATCH * HEADS * NQ * DH * 2);
  _Float16* K16    = (_Float16*)take((size_t)BATCH * HEADS * TTP * DH * 2);
  _Float16* Vt16   = (_Float16*)take((size_t)BATCH * HEADS * DH * TTP * 2);
  _Float16* Kip16  = (_Float16*)take((size_t)BATCH * HEADS * TI * DH * 2);
  _Float16* VipT16 = (_Float16*)take((size_t)BATCH * HEADS * DH * TI * 2);
  _Float16* relk16 = (_Float16*)take((size_t)48 * 64 * 2);
  _Float16* relvT16= (_Float16*)take((size_t)64 * 64 * 2);

  const int nx = BATCH * NQ * DM;
  const int nc = BATCH * CTXLEN * DM;
  cvt_f16<<<(nx + 255) / 256, 256, 0, stream>>>(x, xob16, nx);
  cvt_f16<<<(nc + 255) / 256, 256, 0, stream>>>(ctx, ctx16, nc);
  transpose_w16<<<4096, 256, 0, stream>>>(Wq,    WqT);
  transpose_w16<<<4096, 256, 0, stream>>>(Wk,    WkT);
  transpose_w16<<<4096, 256, 0, stream>>>(Wv,    WvT);
  transpose_w16<<<4096, 256, 0, stream>>>(Wk_ip, WkipT);
  transpose_w16<<<4096, 256, 0, stream>>>(Wv_ip, WvipT);
  transpose_w16<<<4096, 256, 0, stream>>>(Wout,  WoT);
  prep_rel<<<16, 256, 0, stream>>>(rel_k, rel_v, relk16, relvT16);

  dim3 blk(32, 8);
  proj_q<<<dim3(2048, 2), blk, 0, stream>>>(xob16, WqT, Q16);
  proj_ctx<<<dim3(TTP / 16, 2, BATCH), blk, 0, stream>>>(ctx16, 0,  TT, TTP, WkT,   K16,   0);
  proj_ctx<<<dim3(TTP / 16, 2, BATCH), blk, 0, stream>>>(ctx16, 0,  TT, TTP, WvT,   Vt16,  1);
  proj_ctx<<<dim3(TI  / 16, 2, BATCH), blk, 0, stream>>>(ctx16, TT, TI, TI,  WkipT, Kip16, 0);
  proj_ctx<<<dim3(TI  / 16, 2, BATCH), blk, 0, stream>>>(ctx16, TT, TI, TI,  WvipT, VipT16, 1);

  attn_kernel<<<dim3(NQ / 16, HEADS, BATCH), 32, 0, stream>>>(
      Q16, K16, Vt16, Kip16, VipT16, relk16, relvT16, alpha, xob16);

  out_proj<<<dim3(2048, 2), blk, 0, stream>>>(xob16, WoT, bout, y);
}
